// RRGNN_90941637525590
// MI455X (gfx1250) — matrix operand order repaired and checked by reading, not compiled, not run on hardware
//
#include <hip/hip_runtime.h>
#include <hip/hip_bf16.h>

#define DIN  128
#define DH   128
#define DOUT 64

typedef __attribute__((ext_vector_type(16))) __bf16 v16bf;
typedef __attribute__((ext_vector_type(8)))  float  v8f;

__device__ __forceinline__ unsigned short f2bf(float f) {
  unsigned int u = __builtin_bit_cast(unsigned int, f);
  u += 0x7FFFu + ((u >> 16) & 1u);          // round-to-nearest-even
  return (unsigned short)(u >> 16);
}

// ---------------- utility ----------------
__global__ void k_zero_f4(float4* p, long n4) {
  long i = (long)blockIdx.x * blockDim.x + threadIdx.x;
  if (i < n4) p[i] = make_float4(0.f, 0.f, 0.f, 0.f);
}

__global__ void k_degree(const int* __restrict__ dst, float* __restrict__ deg, int E) {
  int i = blockIdx.x * blockDim.x + threadIdx.x;
  if (i < E) atomicAdd(&deg[dst[i]], 1.0f);
}

// One wave per edge: 32 lanes x float4 = 128 floats (coalesced row read),
// f32 atomic adds into destination row (L2-resident on MI455X: 192MB L2).
__global__ void k_scatter(const float* __restrict__ h, const int* __restrict__ src,
                          const int* __restrict__ dst, float* __restrict__ agg, int E) {
  long gid = (long)blockIdx.x * blockDim.x + threadIdx.x;
  int e = (int)(gid >> 5);
  if (e >= E) return;
  int c = ((int)gid & 31) << 2;
  int s = src[e], d = dst[e];
  float4 v = *(const float4*)(h + (long)s * DIN + c);
  float* a = agg + (long)d * DIN + c;
  atomicAdd(a + 0, v.x); atomicAdd(a + 1, v.y);
  atomicAdd(a + 2, v.z); atomicAdd(a + 3, v.w);
}

// Build bf16 combined activation matrix Hc[n] = [ agg[n]/max(deg,1) | h[n] ]  (K = 256)
__global__ void k_combine(const float* __restrict__ h, const float* __restrict__ agg,
                          const float* __restrict__ deg, unsigned short* __restrict__ Hc,
                          int nrows) {
  long gid = (long)blockIdx.x * blockDim.x + threadIdx.x;
  if (gid >= (long)nrows * DIN) return;
  int n = (int)(gid >> 7), c = (int)gid & 127;
  float inv = 1.0f / fmaxf(deg[n], 1.0f);
  unsigned short* row = Hc + (long)n * (2 * DIN);
  row[c]       = f2bf(agg[gid] * inv);
  row[DIN + c] = f2bf(h[gid]);
}

// Pack weights into N-major bf16: Bp[o][k] = (k<128 ? Wl[o][k] : Wr[o][k-128])
__global__ void k_prepw(const float* __restrict__ Wl, const float* __restrict__ Wr,
                        unsigned short* __restrict__ Bp, int dout) {
  int gid = blockIdx.x * blockDim.x + threadIdx.x;
  int total = 2 * DIN * dout;
  if (gid >= total) return;
  int o = gid / (2 * DIN), k = gid - o * (2 * DIN);
  float w = (k < DIN) ? Wl[(long)o * DIN + k] : Wr[(long)o * DIN + (k - DIN)];
  Bp[gid] = f2bf(w);
}

// out[N,dout] = Hc[N,256] @ Bp[256,dout] + bias.  One wave per 16x16 tile,
// v_wmma_f32_16x16x32_bf16 over 8 K-steps, fragments per ISA layout tables.
__global__ void k_gemm_wmma(const unsigned short* __restrict__ Hc,
                            const unsigned short* __restrict__ Bp,
                            const float* __restrict__ bias,
                            float* __restrict__ out, int nrows, int dout) {
  const int K = 2 * DIN;                       // 256
  int wave = threadIdx.x >> 5;
  int lane = threadIdx.x & 31;
  int ntiles = dout >> 4;
  int tile = blockIdx.x * (blockDim.x >> 5) + wave;
  int mt = tile / ntiles, nt = tile - mt * ntiles;
  if (mt * 16 >= nrows) return;
  int half = lane >> 4, l15 = lane & 15;
  int n0 = nt * 16;
  const unsigned short* arow = Hc + (long)(mt * 16 + l15) * K;
  v8f c = {};
#pragma unroll
  for (int k0 = 0; k0 < K; k0 += 32) {
    union { uint4 u[2]; v16bf v; } A, B;
    // A 16x32 bf16: lane half 0 -> K {k0..k0+7, k0+16..k0+23}; half 1 -> +8
    A.u[0] = *(const uint4*)(arow + k0 + half * 8);
    A.u[1] = *(const uint4*)(arow + k0 + 16 + half * 8);
    // B 32x16 bf16: lane = N column; half 0 -> K {k0..k0+7, k0+16..k0+23}; half 1 -> +8
    const unsigned short* brow = Bp + (long)(n0 + l15) * K + k0 + half * 8;
    B.u[0] = *(const uint4*)(brow);
    B.u[1] = *(const uint4*)(brow + 16);
    c = __builtin_amdgcn_wmma_f32_16x16x32_bf16(false, A.v, false, B.v,
                                                (short)0, c, false, false);
  }
  float b = bias[n0 + l15];
  // C/D layout: VGPR r -> row (half*8 + r), col = lane&15
  float* crow = out + (long)(mt * 16 + half * 8) * dout + n0 + l15;
#pragma unroll
  for (int r = 0; r < 8; ++r) crow[(long)r * dout] = c[r] + b;
}

// Per-column partial sums for BN: block of 128 threads (thread = column) over 256 rows
__global__ void k_bn_stats(const float* __restrict__ h, float* __restrict__ stats, int nrows) {
  int col = threadIdx.x;
  int r0 = blockIdx.x * 256;
  int rend = min(r0 + 256, nrows);
  float s = 0.f, s2 = 0.f;
  for (int r = r0; r < rend; ++r) {
    float v = h[(long)r * DH + col];
    s += v; s2 += v * v;
  }
  atomicAdd(&stats[col], s);
  atomicAdd(&stats[DH + col], s2);
}

__global__ void k_bn_relu(const float* __restrict__ in, const float* __restrict__ stats,
                          const float* __restrict__ g, const float* __restrict__ b,
                          float* __restrict__ outp, int nrows) {
  long gid = (long)blockIdx.x * blockDim.x + threadIdx.x;
  if (gid >= (long)nrows * DH) return;
  int c = (int)gid & 127;
  float invn = 1.0f / (float)nrows;
  float mu  = stats[c] * invn;
  float var = stats[DH + c] * invn - mu * mu;
  float y = (in[gid] - mu) * rsqrtf(var + 1e-5f) * g[c] + b[c];
  outp[gid] = fmaxf(y, 0.0f);
}

// Row softmax over 64 logits: one wave per row, 2 values/lane, shfl_xor reductions
__global__ void k_softmax64(const float* __restrict__ in, float* __restrict__ out, int nrows) {
  int wave = threadIdx.x >> 5, lane = threadIdx.x & 31;
  int row = blockIdx.x * (blockDim.x >> 5) + wave;
  if (row >= nrows) return;
  const float* r = in + (long)row * DOUT;
  float a = r[lane], b = r[lane + 32];
  float m = fmaxf(a, b);
#pragma unroll
  for (int off = 16; off; off >>= 1) m = fmaxf(m, __shfl_xor(m, off, 32));
  float ea = __expf(a - m), eb = __expf(b - m);
  float s = ea + eb;
#pragma unroll
  for (int off = 16; off; off >>= 1) s += __shfl_xor(s, off, 32);
  float inv = 1.0f / s;
  float* o = out + (long)row * DOUT;
  o[lane] = ea * inv; o[lane + 32] = eb * inv;
}

// ---------------- driver ----------------
extern "C" void kernel_launch(void* const* d_in, const int* in_sizes, int n_in,
                              void* d_out, int out_size, void* d_ws, size_t ws_size,
                              hipStream_t stream) {
  const float* x   = (const float*)d_in[0];
  const int*   ei  = (const int*)  d_in[1];
  const float* Wl0 = (const float*)d_in[2];
  const float* bl0 = (const float*)d_in[3];
  const float* Wr0 = (const float*)d_in[4];
  const float* Wl1 = (const float*)d_in[5];
  const float* bl1 = (const float*)d_in[6];
  const float* Wr1 = (const float*)d_in[7];
  const float* Wl2 = (const float*)d_in[8];
  const float* bl2 = (const float*)d_in[9];
  const float* Wr2 = (const float*)d_in[10];
  const float* g0  = (const float*)d_in[11];
  const float* be0 = (const float*)d_in[12];
  const float* g1  = (const float*)d_in[13];
  const float* be1 = (const float*)d_in[14];

  const int N = in_sizes[0] / DIN;
  const int E = in_sizes[1] / 2;
  const int* src = ei;
  const int* dst = ei + E;

  auto al = [](size_t v) { return (v + 255) & ~(size_t)255; };
  char* p = (char*)d_ws;
  float* deg = (float*)p;                    p += al((size_t)N * 4);
  float* agg = (float*)p;                    p += al((size_t)N * DH * 4);  // reused as GEMM out
  float* h1  = (float*)p;                    p += al((size_t)N * DH * 4);
  unsigned short* Hc = (unsigned short*)p;   p += al((size_t)N * 2 * DIN * 2);
  unsigned short* Bp = (unsigned short*)p;   p += al((size_t)2 * DIN * DH * 2);
  float* stats = (float*)p;                  p += al(256 * 4);

  const int TB = 256;
  long nND = (long)N * DH;                    // 6.4M
  int bND4   = (int)((nND / 4 + TB - 1) / TB);
  int bDeg4  = (int)((N / 4 + TB - 1) / TB);
  int bE     = (E + TB - 1) / TB;
  int bScat  = (int)(((long)E * 32 + TB - 1) / TB);
  int bND    = (int)((nND + TB - 1) / TB);
  int bPW    = (2 * DIN * DH + TB - 1) / TB;
  int bStats = (N + 255) / 256;

  // degree (shared by all layers)
  k_zero_f4<<<bDeg4, TB, 0, stream>>>((float4*)deg, N / 4);
  k_degree<<<bE, TB, 0, stream>>>(dst, deg, E);

  auto sage = [&](const float* hin, const float* Wl, const float* bl, const float* Wr,
                  int dout, float* gout) {
    k_zero_f4<<<bND4, TB, 0, stream>>>((float4*)agg, nND / 4);
    k_scatter<<<bScat, TB, 0, stream>>>(hin, src, dst, agg, E);
    k_combine<<<bND, TB, 0, stream>>>(hin, agg, deg, Hc, N);
    k_prepw<<<bPW, TB, 0, stream>>>(Wl, Wr, Bp, dout);
    int tiles = ((N + 15) / 16) * (dout / 16);
    k_gemm_wmma<<<(tiles + 7) / 8, TB, 0, stream>>>(Hc, Bp, bl, gout, N, dout);
  };

  // Layer 0: SAGE -> BN -> ReLU
  sage(x, Wl0, bl0, Wr0, DH, agg);
  k_zero_f4<<<1, 64, 0, stream>>>((float4*)stats, 64);
  k_bn_stats<<<bStats, DH, 0, stream>>>(agg, stats, N);
  k_bn_relu<<<bND, TB, 0, stream>>>(agg, stats, g0, be0, h1, N);

  // Layer 1: SAGE -> BN -> ReLU
  sage(h1, Wl1, bl1, Wr1, DH, agg);
  k_zero_f4<<<1, 64, 0, stream>>>((float4*)stats, 64);
  k_bn_stats<<<bStats, DH, 0, stream>>>(agg, stats, N);
  k_bn_relu<<<bND, TB, 0, stream>>>(agg, stats, g1, be1, h1, N);

  // Layer 2: SAGE -> softmax
  sage(h1, Wl2, bl2, Wr2, DOUT, agg);
  k_softmax64<<<(N + 7) / 8, TB, 0, stream>>>(agg, (float*)d_out, N);

  (void)n_in; (void)out_size; (void)ws_size;
}
